// FeatureAlignment_37795712204915
// MI455X (gfx1250) — compile-verified, matrix operand order checked by reading, not run on hardware
//
#include <hip/hip_runtime.h>
#include <stdint.h>

typedef __attribute__((ext_vector_type(16))) __bf16    v16bf;
typedef __attribute__((ext_vector_type(8)))  float     v8f;
typedef __attribute__((ext_vector_type(4)))  unsigned  uint4v;
typedef __attribute__((ext_vector_type(4)))  int       v4i;

#define H_    160
#define W_    160
#define HW_   (160 * 160)
#define B_    8
#define C_    64
#define O_    64
#define KTOT  576            // C * 9
#define NCHNK 18             // 576 / 32

#if __has_builtin(__builtin_amdgcn_global_load_async_to_lds_b128)
#define HAVE_ASYNC_LDS 1
typedef __attribute__((address_space(1))) v4i GV4;   // global (__device__) int4
typedef __attribute__((address_space(3))) v4i LV4;   // LDS (__shared__) int4
#endif

union BfVec {
  v16bf    v;
  uint16_t u[16];
  uint4v   q[2];
};

__device__ __forceinline__ uint16_t f2bf(float f) {
  unsigned u = __float_as_uint(f);
  unsigned r = u + 0x7fffu + ((u >> 16) & 1u);   // round-to-nearest-even
  return (uint16_t)(r >> 16);
}

// packed 2x bf16 fma on raw 32-bit pairs: d = a*b + c  (VOP3P V_PK_FMA_BF16)
__device__ __forceinline__ unsigned pk_fma_bf16(unsigned a, unsigned b, unsigned c) {
  unsigned d;
  asm("v_pk_fma_bf16 %0, %1, %2, %3" : "=v"(d) : "v"(a), "v"(b), "v"(c));
  return d;
}

// ---------------- K0a: x (NCHW f32) -> xb (NHWC bf16) ----------------
__global__ __launch_bounds__(256) void k_pack_x(const float* __restrict__ x,
                                                uint16_t* __restrict__ xb) {
  const int i   = blockIdx.x * 256 + threadIdx.x;   // NHWC flat index
  const int c   = i & 63;
  const int pix = i >> 6;
  const int b   = pix / HW_;
  const int hw  = pix - b * HW_;
  xb[i] = f2bf(x[(b * C_ + c) * HW_ + hw]);
}

// ---------------- K0b: repack weights, K = tap*64 + c, bf16 ----------------
__global__ __launch_bounds__(256) void k_pack_w(const float* __restrict__ w_dcn,
                                                const float* __restrict__ w_off,
                                                uint16_t* __restrict__ wdcnp,
                                                uint16_t* __restrict__ woffp) {
  const int i = blockIdx.x * 256 + threadIdx.x;     // 0 .. 96*576-1
  if (i < O_ * KTOT) {
    const int o = i / KTOT, k = i - o * KTOT;
    const int t = k >> 6, c = k & 63;
    wdcnp[i] = f2bf(w_dcn[(o * C_ + c) * 9 + t]);
  } else {
    const int j = i - O_ * KTOT;                    // 32 padded rows for w_off
    const int o = j / KTOT, k = j - o * KTOT;
    const int t = k >> 6, c = k & 63;
    woffp[j] = (o < 18) ? f2bf(w_off[(o * C_ + c) * 9 + t]) : (uint16_t)0;
  }
}

// ---------------- K1: offset conv (3x3, 64->18) via WMMA bf16 ----------------
__global__ __launch_bounds__(256) void k_off_conv(const uint16_t* __restrict__ xb,
                                                  const uint16_t* __restrict__ woffp,
                                                  const float* __restrict__ b_off,
                                                  float* __restrict__ offs) {
  const int lane = threadIdx.x & 31;
  const int wid  = threadIdx.x >> 5;
  const int g    = lane >> 4;     // K-octet group
  const int n    = lane & 15;     // column (pixel) / A row
  const int pix  = blockIdx.x * 128 + wid * 16 + n;
  const int b    = pix / HW_;
  const int hw   = pix - b * HW_;
  const int y    = hw / W_;
  const int x    = hw - y * W_;

  const uint4v zero = {};
  v8f acc0 = {};
  v8f acc1 = {};

  for (int kc = 0; kc < NCHNK; ++kc) {
    const int tap = kc >> 1;
    const int cb  = (kc & 1) * 32;
    const int iy  = y + (tap / 3) - 1;
    const int ix  = x + (tap % 3) - 1;

    BfVec bv;
    if (iy >= 0 && iy < H_ && ix >= 0 && ix < W_) {
      const uint16_t* src = xb + ((((b * H_ + iy) * W_) + ix) << 6) + cb;
      bv.q[0] = *(const uint4v*)(src + g * 8);
      bv.q[1] = *(const uint4v*)(src + 16 + g * 8);
    } else {
      bv.q[0] = zero;
      bv.q[1] = zero;
    }

    const uint16_t* wr0 = woffp + (0 * 16 + n) * KTOT + kc * 32;
    const uint16_t* wr1 = woffp + (1 * 16 + n) * KTOT + kc * 32;
    BfVec a0, a1;
    a0.q[0] = *(const uint4v*)(wr0 + g * 8);
    a0.q[1] = *(const uint4v*)(wr0 + 16 + g * 8);
    a1.q[0] = *(const uint4v*)(wr1 + g * 8);
    a1.q[1] = *(const uint4v*)(wr1 + 16 + g * 8);

    acc0 = __builtin_amdgcn_wmma_f32_16x16x32_bf16(false, a0.v, false, bv.v,
                                                   (short)0, acc0, false, false);
    acc1 = __builtin_amdgcn_wmma_f32_16x16x32_bf16(false, a1.v, false, bv.v,
                                                   (short)0, acc1, false, false);
  }

#pragma unroll
  for (int r = 0; r < 8; ++r) {
    const int o0 = r + g * 8;            // 0..15, always valid
    offs[pix * 18 + o0] = acc0[r] + b_off[o0];
    const int o1 = 16 + r + g * 8;       // 16..31, keep 16,17
    if (o1 < 18) offs[pix * 18 + o1] = acc1[r] + b_off[o1];
  }
}

// bilinear-blend 8 consecutive channels from 4 corners; packed bf16 fma chain.
// wXXp are pre-splatted bf16 pairs; result is 4 dwords = 8 bf16 values.
__device__ __forceinline__ uint4v sample8p(const uint16_t* __restrict__ xb,
                                           unsigned a00, unsigned a01,
                                           unsigned a10, unsigned a11,
                                           unsigned w00p, unsigned w01p,
                                           unsigned w10p, unsigned w11p) {
  const uint4v q00 = *(const uint4v*)(xb + a00);
  const uint4v q01 = *(const uint4v*)(xb + a01);
  const uint4v q10 = *(const uint4v*)(xb + a10);
  const uint4v q11 = *(const uint4v*)(xb + a11);
  uint4v r;
#pragma unroll
  for (int p = 0; p < 4; ++p) {
    unsigned t = pk_fma_bf16(q11[p], w11p, 0u);
    t = pk_fma_bf16(q10[p], w10p, t);
    t = pk_fma_bf16(q01[p], w01p, t);
    r[p] = pk_fma_bf16(q00[p], w00p, t);
  }
  return r;
}

// ---------------- K2: deform sample + 64x576 GEMM + BN + ReLU ----------------
__global__ __launch_bounds__(256) void k_dcn(const uint16_t* __restrict__ xb,
                                             const uint16_t* __restrict__ wdcnp,
                                             const float* __restrict__ offs,
                                             const float* __restrict__ b_dcn,
                                             const float* __restrict__ gamma,
                                             const float* __restrict__ beta,
                                             const float* __restrict__ rmean,
                                             const float* __restrict__ rvar,
                                             float* __restrict__ out) {
  __shared__ unsigned s_w[128 * 9 * 4];    // splatted bf16-pair weights (18.4 KB)
  __shared__ unsigned s_a[128 * 9 * 4];    // corner NHWC elem offsets   (18.4 KB)
  __shared__ uint16_t s_wgt[O_ * KTOT];    // staged A matrix            (73.7 KB)

  const int pixBase = blockIdx.x * 128;

  // ---- stage the 64x576 bf16 weight matrix into LDS (async-to-LDS path) ----
#ifdef HAVE_ASYNC_LDS
  for (int i = threadIdx.x; i < (O_ * KTOT) / 8; i += 256) {
    __builtin_amdgcn_global_load_async_to_lds_b128(
        (GV4*)(wdcnp + i * 8), (LV4*)(s_wgt + i * 8), 0, 0);
  }
  asm volatile("s_wait_asynccnt 0" ::: "memory");
#else
  for (int i = threadIdx.x; i < (O_ * KTOT) / 8; i += 256) {
    *(uint4v*)(s_wgt + i * 8) = *(const uint4v*)(wdcnp + i * 8);
  }
#endif

  // ---- precompute per-(pixel, tap) corner weights (bf16 pair) + addresses ----
  for (int i = threadIdx.x; i < 128 * 9; i += 256) {
    const int pl  = i / 9;
    const int tap = i - pl * 9;
    const int pix = pixBase + pl;
    const int b   = pix / HW_;
    const int hw  = pix - b * HW_;
    const int y   = hw / W_;
    const int x   = hw - y * W_;
    const float oy = offs[pix * 18 + tap * 2 + 0];
    const float ox = offs[pix * 18 + tap * 2 + 1];
    const float py = (float)(y + tap / 3 - 1) + oy;
    const float px = (float)(x + tap % 3 - 1) + ox;
    const float fy = floorf(py), fx = floorf(px);
    const int   y0 = (int)fy,   x0 = (int)fx;
    const float wy1 = py - fy, wy0 = 1.0f - wy1;
    const float wx1 = px - fx, wx0 = 1.0f - wx1;
    const bool vy0 = (y0 >= 0)  && (y0 <= H_ - 1);
    const bool vy1 = (y0 >= -1) && (y0 <= H_ - 2);
    const bool vx0 = (x0 >= 0)  && (x0 <= W_ - 1);
    const bool vx1 = (x0 >= -1) && (x0 <= W_ - 2);
    const int cy0 = min(max(y0, 0), H_ - 1);
    const int cy1 = min(max(y0 + 1, 0), H_ - 1);
    const int cx0 = min(max(x0, 0), W_ - 1);
    const int cx1 = min(max(x0 + 1, 0), W_ - 1);
    const unsigned rb = (unsigned)(b * H_);
    const int t4 = i * 4;
    const unsigned h0 = f2bf((vy0 && vx0) ? wy0 * wx0 : 0.0f);
    const unsigned h1 = f2bf((vy0 && vx1) ? wy0 * wx1 : 0.0f);
    const unsigned h2 = f2bf((vy1 && vx0) ? wy1 * wx0 : 0.0f);
    const unsigned h3 = f2bf((vy1 && vx1) ? wy1 * wx1 : 0.0f);
    s_w[t4 + 0] = (h0 << 16) | h0;
    s_w[t4 + 1] = (h1 << 16) | h1;
    s_w[t4 + 2] = (h2 << 16) | h2;
    s_w[t4 + 3] = (h3 << 16) | h3;
    s_a[t4 + 0] = (((rb + cy0) * W_) + cx0) << 6;
    s_a[t4 + 1] = (((rb + cy0) * W_) + cx1) << 6;
    s_a[t4 + 2] = (((rb + cy1) * W_) + cx0) << 6;
    s_a[t4 + 3] = (((rb + cy1) * W_) + cx1) << 6;
  }
  __syncthreads();

  const int lane   = threadIdx.x & 31;
  const int wid    = threadIdx.x >> 5;
  const int g      = lane >> 4;
  const int n      = lane & 15;
  const int plocal = wid * 16 + n;
  const int pix    = pixBase + plocal;
  const int b      = pix / HW_;
  const int hw     = pix - b * HW_;

  v8f acc[4] = {};

  for (int kc = 0; kc < NCHNK; ++kc) {
    const int tap = kc >> 1;
    const int cb  = (kc & 1) * 32;
    const int t4  = (plocal * 9 + tap) * 4;
    const unsigned w00p = s_w[t4 + 0], w01p = s_w[t4 + 1];
    const unsigned w10p = s_w[t4 + 2], w11p = s_w[t4 + 3];
    const unsigned a00 = s_a[t4 + 0], a01 = s_a[t4 + 1];
    const unsigned a10 = s_a[t4 + 2], a11 = s_a[t4 + 3];

    const int cA = cb + g * 8;
    const int cB = cb + 16 + g * 8;

    BfVec bv;
    bv.q[0] = sample8p(xb, a00 + cA, a01 + cA, a10 + cA, a11 + cA,
                       w00p, w01p, w10p, w11p);
    bv.q[1] = sample8p(xb, a00 + cB, a01 + cB, a10 + cB, a11 + cB,
                       w00p, w01p, w10p, w11p);

#pragma unroll
    for (int m = 0; m < 4; ++m) {
      const uint16_t* wr = s_wgt + (m * 16 + n) * KTOT + kc * 32;   // ds_load_b128
      BfVec av;
      av.q[0] = *(const uint4v*)(wr + g * 8);
      av.q[1] = *(const uint4v*)(wr + 16 + g * 8);
      acc[m] = __builtin_amdgcn_wmma_f32_16x16x32_bf16(false, av.v, false, bv.v,
                                                       (short)0, acc[m], false, false);
    }
  }

#pragma unroll
  for (int m = 0; m < 4; ++m) {
#pragma unroll
    for (int r = 0; r < 8; ++r) {
      const int o = m * 16 + g * 8 + r;
      const float inv = gamma[o] * rsqrtf(rvar[o] + 1e-5f);
      const float sh  = beta[o] - rmean[o] * inv;
      const float z   = acc[m][r] + b_dcn[o];
      out[(b * O_ + o) * HW_ + hw] = fmaxf(fmaf(z, inv, sh), 0.0f);
    }
  }
}

extern "C" void kernel_launch(void* const* d_in, const int* in_sizes, int n_in,
                              void* d_out, int out_size, void* d_ws, size_t ws_size,
                              hipStream_t stream) {
  const float* x     = (const float*)d_in[0];
  const float* w_off = (const float*)d_in[1];
  const float* b_off = (const float*)d_in[2];
  const float* w_dcn = (const float*)d_in[3];
  const float* b_dcn = (const float*)d_in[4];
  const float* gamma = (const float*)d_in[5];
  const float* beta  = (const float*)d_in[6];
  const float* rmean = (const float*)d_in[7];
  const float* rvar  = (const float*)d_in[8];
  float* out = (float*)d_out;

  char* ws = (char*)d_ws;
  // xb:    8*160*160*64 bf16            = 26,214,400 B
  // offs:  8*160*160*18 f32             = 14,745,600 B
  // wdcnp: 64*576 bf16                  =     73,728 B
  // woffp: 32*576 bf16                  =     36,864 B
  uint16_t* xb    = (uint16_t*)(ws);
  float*    offs  = (float*)(ws + 26214400);
  uint16_t* wdcnp = (uint16_t*)(ws + 26214400 + 14745600);
  uint16_t* woffp = (uint16_t*)(ws + 26214400 + 14745600 + 73728);

  k_pack_x<<<51200, 256, 0, stream>>>(x, xb);                       // 13.1M elems
  k_pack_w<<<216, 256, 0, stream>>>(w_dcn, w_off, wdcnp, woffp);    // 55,296 elems
  k_off_conv<<<1600, 256, 0, stream>>>(xb, woffp, b_off, offs);     // 128 px / block
  k_dcn<<<1600, 256, 0, stream>>>(xb, wdcnp, offs, b_dcn,
                                  gamma, beta, rmean, rvar, out);   // 128 px / block
}
